// SetAbstractionModule_87050397155548
// MI455X (gfx1250) — matrix operand order repaired and checked by reading, not compiled, not run on hardware
//
#include <hip/hip_runtime.h>

// ---- problem constants (match reference) ----
#define B_   16
#define N_   4096
#define C_   64
#define M_   1024
#define K_   32
#define RAD2 (0.2f * 0.2f)
#define EPS_ 1e-5f

// ---- CDNA5 WMMA vector types ----
typedef __attribute__((ext_vector_type(16))) __bf16 v16bf;
typedef __attribute__((ext_vector_type(8)))  __bf16 v8bf;
typedef __attribute__((ext_vector_type(8)))  float  v8f;

// Fragment table in workspace (bf16, 512 elems = 1 KB each):
//   layer1: f = kt*4 + nt        (kt:0..2, nt:0..3)  -> f in [0,12)
//   layer2: f = 12 + kt*4 + nt   (kt:0..1, nt:0..3)  -> f in [12,20)
//   layer3: f = 20 + kt*8 + nt   (kt:0..1, nt:0..7)  -> f in [20,36)
#define NFRAG 36
#define FRAG_BYTES (NFRAG * 512 * 2)

// =====================================================================
// Kernel 1: farthest point sampling, one block (1024 thr) per batch.
// Emits centroids [B,3,M] directly into d_out (matches reference: first
// emitted index is 0; argmax ties resolve to the lowest index).
// =====================================================================
__global__ __launch_bounds__(1024) void fps_kernel(
    const float* __restrict__ points, float* __restrict__ cent)
{
  __shared__ float sx[N_], sy[N_], sz[N_];
  __shared__ unsigned long long skey[32];
  __shared__ int sLast;

  const int b   = blockIdx.x;
  const int tid = threadIdx.x;
  const float* px = points + (size_t)b * 3 * N_;
  const float* py = px + N_;
  const float* pz = py + N_;

  float lx[4], ly[4], lz[4], ld[4];
#pragma unroll
  for (int j = 0; j < 4; ++j) {
    int n = tid + j * 1024;
    float x = px[n], y = py[n], z = pz[n];
    sx[n] = x; sy[n] = y; sz[n] = z;
    lx[j] = x; ly[j] = y; lz[j] = z; ld[j] = 1e10f;
  }
  __syncthreads();

  int last = 0;
  for (int m = 0; m < M_; ++m) {
    if (tid == 0) {
      cent[(b * 3 + 0) * M_ + m] = sx[last];
      cent[(b * 3 + 1) * M_ + m] = sy[last];
      cent[(b * 3 + 2) * M_ + m] = sz[last];
    }
    const float qx = sx[last], qy = sy[last], qz = sz[last];
    unsigned long long best = 0ull;
#pragma unroll
    for (int j = 0; j < 4; ++j) {
      int n = tid + j * 1024;
      float dx = lx[j] - qx, dy = ly[j] - qy, dz = lz[j] - qz;
      float d = dx * dx + dy * dy + dz * dz;
      d = fminf(ld[j], d);
      ld[j] = d;
      unsigned long long key =
          ((unsigned long long)__float_as_uint(d) << 32) |
          (unsigned)(0xFFFFFFFFu - (unsigned)n);
      best = key > best ? key : best;
    }
#pragma unroll
    for (int off = 16; off; off >>= 1) {
      unsigned long long o = __shfl_xor(best, off, 32);
      best = o > best ? o : best;
    }
    if ((tid & 31) == 0) skey[tid >> 5] = best;
    __syncthreads();
    if (tid < 32) {
      unsigned long long k = skey[tid];
#pragma unroll
      for (int off = 16; off; off >>= 1) {
        unsigned long long o = __shfl_xor(k, off, 32);
        k = o > k ? o : k;
      }
      if (tid == 0) sLast = (int)(0xFFFFFFFFu - (unsigned)k);
    }
    __syncthreads();
    last = sLast;
  }
}

// =====================================================================
// Kernel 2: fold BN scale into weights, build bf16 WMMA B-fragments
// (dense 16-bit B layout: lane<16 -> K 0..7,16..23, lane>=16 -> K 8..15,24..31,
//  2 K-rows per VGPR slot) + folded biases, into workspace.
// =====================================================================
__global__ void prep_kernel(
    const float* __restrict__ w1, const float* __restrict__ b1,
    const float* __restrict__ g1, const float* __restrict__ be1,
    const float* __restrict__ w2, const float* __restrict__ b2,
    const float* __restrict__ g2, const float* __restrict__ be2,
    const float* __restrict__ w3, const float* __restrict__ b3,
    const float* __restrict__ g3, const float* __restrict__ be3,
    unsigned short* __restrict__ frags_u, float* __restrict__ biasp)
{
  __bf16* frags = reinterpret_cast<__bf16*>(frags_u);
  const float inv = 1.0f / sqrtf(1.0f + EPS_);
  const int f = blockIdx.x;
  const int lane = threadIdx.x;

  if (f < NFRAG) {
    const float* w; const float* g; int Kin, kt, nt;
    if (f < 12)      { w = w1; g = g1; Kin = 3 + C_; kt = f / 4;        nt = f % 4; }
    else if (f < 20) { w = w2; g = g2; Kin = 64;     kt = (f - 12) / 4; nt = (f - 12) % 4; }
    else             { w = w3; g = g3; Kin = 64;     kt = (f - 20) / 8; nt = (f - 20) % 8; }
    const int nloc  = lane & 15;
    const int khalf = (lane & 16);        // 0 or 16
    const int o     = nt * 16 + nloc;
    const float s   = g[o] * inv;
    __bf16* out = frags + (size_t)f * 512 + lane * 16;
#pragma unroll
    for (int v = 0; v < 8; ++v)
#pragma unroll
      for (int j = 0; j < 2; ++j) {
        int kg = kt * 32 + (khalf >> 1) + 2 * v + j + ((v >= 4) ? 8 : 0);
        float x = (kg < Kin) ? w[o * Kin + kg] * s : 0.0f;
        out[2 * v + j] = (__bf16)x;
      }
  } else {
    for (int o = lane; o < 64; o += 32) {
      biasp[o]      = b1[o] * (g1[o] * inv) + be1[o];
      biasp[64 + o] = b2[o] * (g2[o] * inv) + be2[o];
    }
    for (int o = lane; o < 128; o += 32)
      biasp[128 + o] = b3[o] * (g3[o] * inv) + be3[o];
  }
}

// ---- WMMA helpers ----
__device__ __forceinline__ v8f splat8(float x) {
  v8f r;
#pragma unroll
  for (int i = 0; i < 8; ++i) r[i] = x;
  return r;
}

// A-fragment from row-major bf16 LDS tile (ISA 16-bit A 16x32 layout).
__device__ __forceinline__ v16bf load_a(const __bf16* base, int stride,
                                        int mt, int kt, int lane) {
  const int row   = mt * 16 + (lane & 15);
  const int kbase = kt * 32 + ((lane & 16) ? 8 : 0);
  const __bf16* p = base + row * stride + kbase;
  v8bf lo = *(const v8bf*)p;          // K kbase .. kbase+7
  v8bf hi = *(const v8bf*)(p + 16);   // K kbase+16 .. kbase+23
  return __builtin_shufflevector(lo, hi, 0, 1, 2, 3, 4, 5, 6, 7,
                                 8, 9, 10, 11, 12, 13, 14, 15);
}

__device__ __forceinline__ v16bf load_b(const __bf16* frags, int f, int lane) {
  return *(const v16bf*)(frags + (size_t)f * 512 + lane * 16);
}

__device__ __forceinline__ v8f wmma_bf16(v16bf a, v16bf b, v8f c) {
  return __builtin_amdgcn_wmma_f32_16x16x32_bf16(
      false, a, false, b, (short)0, c, false, false);
}

// ReLU + write D fragment (row/VGPR, col/lane) as bf16 row-major to LDS.
__device__ __forceinline__ void store_h(__bf16* h, int stride, int mt, int nt,
                                        int lane, v8f c) {
  const int col   = nt * 16 + (lane & 15);
  const int rbase = mt * 16 + ((lane & 16) ? 8 : 0);
#pragma unroll
  for (int r = 0; r < 8; ++r) {
    float v = c[r];
    v = v > 0.0f ? v : 0.0f;
    h[(rbase + r) * stride + col] = (__bf16)v;
  }
}

// =====================================================================
// Kernel 3: ball query + gather + 3-layer WMMA MLP + K-max.
// One wave per centroid, 8 waves / block.
// =====================================================================
__global__ __launch_bounds__(256) void sa_mlp_kernel(
    const float* __restrict__ points,            // [B,3,N]
    const float* __restrict__ features,          // [B,C,N]
    const unsigned short* __restrict__ frags_u,  // bf16 B-fragments
    const float* __restrict__ biasp,             // folded biases
    const float* __restrict__ cent,              // [B,3,M] (from fps)
    float* __restrict__ outf)                    // [B,128,M]
{
  __shared__ __bf16 sX[8][32 * 96];   // per-wave X / h2 buffer
  __shared__ __bf16 sH[8][32 * 64];   // per-wave h1 buffer
  __shared__ int    sIdx[8][32];

  const __bf16* frags = reinterpret_cast<const __bf16*>(frags_u);
  const int wave = threadIdx.x >> 5;
  const int lane = threadIdx.x & 31;
  const int gw = blockIdx.x * 8 + wave;
  const int b = gw / M_;
  const int m = gw % M_;

  __builtin_prefetch(frags_u, 0, 3);  // global_prefetch weights into caches

  const float cx = cent[(b * 3 + 0) * M_ + m];
  const float cy = cent[(b * 3 + 1) * M_ + m];
  const float cz = cent[(b * 3 + 2) * M_ + m];

  const float* px = points + (size_t)b * 3 * N_;
  const float* py = px + N_;
  const float* pz = py + N_;

  // ---- ball query: first K in-radius points by index (wave-cooperative) ----
  int found = 0;
  for (int base = 0; base < N_ && found < K_; base += 32) {
    const int n = base + lane;
    float dx = px[n] - cx, dy = py[n] - cy, dz = pz[n] - cz;
    float d2 = dx * dx + dy * dy + dz * dz;
    bool hit = d2 <= RAD2;
    unsigned msk = (unsigned)__ballot(hit);
    int before = __popc(msk & ((1u << lane) - 1u));
    int pos = found + before;
    if (hit && pos < K_) sIdx[wave][pos] = n;
    found += __popc(msk);
  }
  if (found > K_) found = K_;
  if (found == 0) {
    sIdx[wave][lane] = N_ - 1;  // degenerate case (matches reference clip)
  } else {
    int first = sIdx[wave][0];
    for (int pos = found + lane; pos < K_; pos += 32) sIdx[wave][pos] = first;
  }
  __syncthreads();

  // ---- gather: row 'lane' of X = [local xyz | 64 feats | zero pad] ----
  __bf16* X = sX[wave];
  const int nk = sIdx[wave][lane];
  __bf16* row = X + lane * 96;
  row[0] = (__bf16)(px[nk] - cx);
  row[1] = (__bf16)(py[nk] - cy);
  row[2] = (__bf16)(pz[nk] - cz);
  const float* fb = features + (size_t)b * C_ * N_ + nk;
#pragma unroll 8
  for (int c = 0; c < C_; ++c) row[3 + c] = (__bf16)fb[(size_t)c * N_];
#pragma unroll
  for (int c = 3 + C_; c < 96; ++c) row[c] = (__bf16)0.0f;
  __syncthreads();

  // ---- layer 1: X(32x96,bf16) @ W1'(96x64) -> h1 (ReLU, bf16 in sH) ----
  __bf16* H = sH[wave];
  const float* bias1 = biasp;
#pragma unroll
  for (int mt = 0; mt < 2; ++mt) {
    v16bf a0 = load_a(X, 96, mt, 0, lane);
    v16bf a1 = load_a(X, 96, mt, 1, lane);
    v16bf a2 = load_a(X, 96, mt, 2, lane);
#pragma unroll
    for (int nt = 0; nt < 4; ++nt) {
      v8f c = splat8(bias1[nt * 16 + (lane & 15)]);
      c = wmma_bf16(a0, load_b(frags, 0 + nt, lane), c);
      c = wmma_bf16(a1, load_b(frags, 4 + nt, lane), c);
      c = wmma_bf16(a2, load_b(frags, 8 + nt, lane), c);
      store_h(H, 64, mt, nt, lane, c);
    }
  }
  __syncthreads();

  // ---- layer 2: h1(32x64) @ W2'(64x64) -> h2 (ReLU, bf16, reuse sX) ----
  {
    v16bf a00 = load_a(H, 64, 0, 0, lane);
    v16bf a01 = load_a(H, 64, 0, 1, lane);
    v16bf a10 = load_a(H, 64, 1, 0, lane);
    v16bf a11 = load_a(H, 64, 1, 1, lane);
    const float* bias2 = biasp + 64;
#pragma unroll
    for (int nt = 0; nt < 4; ++nt) {
      float bv = bias2[nt * 16 + (lane & 15)];
      v8f c0 = splat8(bv);
      c0 = wmma_bf16(a00, load_b(frags, 12 + nt, lane), c0);
      c0 = wmma_bf16(a01, load_b(frags, 16 + nt, lane), c0);
      store_h(X, 64, 0, nt, lane, c0);
      v8f c1 = splat8(bv);
      c1 = wmma_bf16(a10, load_b(frags, 12 + nt, lane), c1);
      c1 = wmma_bf16(a11, load_b(frags, 16 + nt, lane), c1);
      store_h(X, 64, 1, nt, lane, c1);
    }
  }
  __syncthreads();

  // ---- layer 3: h2(32x64) @ W3'(64x128) -> ReLU -> max over 32 rows ----
  {
    v16bf e00 = load_a(X, 64, 0, 0, lane);
    v16bf e01 = load_a(X, 64, 0, 1, lane);
    v16bf e10 = load_a(X, 64, 1, 0, lane);
    v16bf e11 = load_a(X, 64, 1, 1, lane);
    const float* bias3 = biasp + 128;
    float* outp = outf + (size_t)b * 128 * M_ + m;
#pragma unroll
    for (int nt = 0; nt < 8; ++nt) {
      float bv = bias3[nt * 16 + (lane & 15)];
      v8f c0 = splat8(bv);
      v8f c1 = splat8(bv);
      v16bf w0 = load_b(frags, 20 + nt, lane);
      v16bf w1 = load_b(frags, 28 + nt, lane);
      c0 = wmma_bf16(e00, w0, c0);
      c0 = wmma_bf16(e01, w1, c0);
      c1 = wmma_bf16(e10, w0, c1);
      c1 = wmma_bf16(e11, w1, c1);
      float v = 0.0f;  // ReLU folded into the max (relu(max)=max(relu))
#pragma unroll
      for (int i = 0; i < 8; ++i) {
        v = fmaxf(v, c0[i]);
        v = fmaxf(v, c1[i]);
      }
      float o = fmaxf(v, __shfl_xor(v, 16, 32));
      if (lane < 16) outp[(size_t)(nt * 16 + lane) * M_] = o;
    }
  }
}

// =====================================================================
extern "C" void kernel_launch(void* const* d_in, const int* in_sizes, int n_in,
                              void* d_out, int out_size, void* d_ws, size_t ws_size,
                              hipStream_t stream) {
  const float* points   = (const float*)d_in[0];
  const float* features = (const float*)d_in[1];
  const float* w1  = (const float*)d_in[2];
  const float* b1  = (const float*)d_in[3];
  const float* g1  = (const float*)d_in[4];
  const float* be1 = (const float*)d_in[5];
  const float* w2  = (const float*)d_in[6];
  const float* b2  = (const float*)d_in[7];
  const float* g2  = (const float*)d_in[8];
  const float* be2 = (const float*)d_in[9];
  const float* w3  = (const float*)d_in[10];
  const float* b3  = (const float*)d_in[11];
  const float* g3  = (const float*)d_in[12];
  const float* be3 = (const float*)d_in[13];

  float* out      = (float*)d_out;
  float* cent     = out;                       // [B,3,M]
  float* feat_out = out + (size_t)B_ * 3 * M_; // [B,128,M]

  unsigned short* frags = (unsigned short*)d_ws;
  float* biasp = (float*)((char*)d_ws + FRAG_BYTES);

  fps_kernel<<<B_, 1024, 0, stream>>>(points, cent);
  prep_kernel<<<NFRAG + 1, 32, 0, stream>>>(w1, b1, g1, be1,
                                            w2, b2, g2, be2,
                                            w3, b3, g3, be3,
                                            frags, biasp);
  sa_mlp_kernel<<<(B_ * M_) / 8, 256, 0, stream>>>(points, features,
                                                   frags, biasp, cent, feat_out);
}
